// AEVAE_2843268350432
// MI455X (gfx1250) — matrix operand order repaired and checked by reading, not compiled, not run on hardware
//
#include <hip/hip_runtime.h>

typedef __attribute__((ext_vector_type(16))) _Float16 v16h;
typedef __attribute__((ext_vector_type(8)))  _Float16 v8h;
typedef __attribute__((ext_vector_type(8)))  float    v8f;
typedef __attribute__((ext_vector_type(4)))  unsigned int u32x4;
typedef __attribute__((ext_vector_type(8)))  unsigned int u32x8;

#define TM 64
#define TN 32
#define TK 64
#define LDA (TK + 8)   /* 72 halves -> 144B row pitch: 16B aligned, 36-bank stride */
#define LDB (TK + 8)
#define NTHR 128

// ---------------------------------------------------------------------------
// Weight pre-pass: f32 [O x K] -> f16 [Opad x Kpad], zero-padded.
// Run once per layer; makes the W tile a guard-free aligned 2-D block so the
// GEMM kernel can fetch it with the Tensor Data Mover.
// ---------------------------------------------------------------------------
__global__ __launch_bounds__(256)
void cvt_pad_f16(const float* __restrict__ W, _Float16* __restrict__ out,
                 int O, int K, int Kpad, int total)
{
  const int gid = blockIdx.x * 256 + threadIdx.x;
  if (gid >= total) return;
  const int o = gid / Kpad;
  const int k = gid - o * Kpad;
  out[gid] = (o < O && k < K) ? (_Float16)W[(long)o * K + k] : (_Float16)0.f;
}

// ---------------------------------------------------------------------------
// Fused (gather -> GEMM -> bias -> optional ELU) spiral convolution.
//   A[r,k] = x[b, idx[v,s], c]   with r=(b,v), k=(s,c), K=S*C, M=B*V
//   out[r,o] = act( sum_k A[r,k] * Wh[o,k] + bias[o] )
// idx==nullptr => dense FC mode. Wh is the padded f16 weight arena entry.
// Block: 4 wave32; 64(M) x 32(N) tile; K chunk 64 => 4 v_wmma per barrier pair.
// W tile staged by TDM (tensor_load_to_lds, wave 0), overlapped with A gather;
// LDS pitch 144B realized via D# pad fields (interval=32dw, amount=4dw).
// ---------------------------------------------------------------------------
__global__ __launch_bounds__(NTHR)
void sconv_wmma(const float* __restrict__ x, const int* __restrict__ idx,
                const _Float16* __restrict__ Wh, const float* __restrict__ bias,
                float* __restrict__ out,
                int Bn, int V, int Vin, int S, int C, int O, int act,
                int cshift, int Kpad)
{
  const int K  = S * C;
  const int M  = Bn * V;
  const int m0 = blockIdx.x * TM;
  const int n0 = blockIdx.y * TN;

  __shared__ _Float16 ldsA[TM * LDA];   // [m][k]
  __shared__ _Float16 ldsB[TN * LDB];   // transposed: [n][k], filled by TDM

  const int tid  = threadIdx.x;
  const int wave = tid >> 5;
  const int lane = tid & 31;
  const int half = lane >> 4;
  const int l15  = lane & 15;

  v8f acc0 = {};
  v8f acc1 = {};

  for (int k0 = 0; k0 < Kpad; k0 += TK) {
    // ---- wave 0: kick TDM load of W tile (32 rows x 64 halves) into ldsB
    if (wave == 0) {
      const unsigned long long ga =
          (unsigned long long)(size_t)(Wh + (size_t)n0 * Kpad + k0);
      const unsigned lds_off = (unsigned)(size_t)(void*)&ldsB[0];
      u32x4 g0;
      g0[0] = 1u;                                   // count=1, no gather
      g0[1] = lds_off;                              // lds_addr (bytes)
      g0[2] = (unsigned)ga;                         // global_addr[31:0]
      g0[3] = (unsigned)((ga >> 32) & 0x01FFFFFFu)  // global_addr[56:32]
            | (2u << 30);                           // type=2 (image)
      u32x8 g1;
      g1[0] = (1u << 16)    // data_size = 1 -> 2 bytes
            | (1u << 20)    // pad_enable
            | (4u << 22)    // pad_interval code 4 -> 32 DWORDs (=128B row)
            | (3u << 25);   // pad_amount  code 3 -> 4 DWORDs  (=16B pad)
      g1[1] = ((unsigned)Kpad & 0xFFFFu) << 16;     // tensor_dim0[15:0]
      g1[2] = ((unsigned)Kpad >> 16)                // tensor_dim0[31:16]
            | (32u << 16);                          // tensor_dim1 = 32 rows
      g1[3] = (64u << 16);                          // tile_dim0 = 64 elems
      g1[4] = 32u;                                  // tile_dim1 = 32, tile_dim2=0
      g1[5] = (unsigned)Kpad;                       // tensor_dim0_stride[31:0]
      g1[6] = 0u;                                   // stride[47:32], dim1_stride lo
      g1[7] = 0u;
      asm volatile("tensor_load_to_lds %0, %1" :: "s"(g0), "s"(g1) : "memory");
    }

    // ---- all waves: stage gathered A chunk (64 x 64), f32 -> f16, zero-pad.
    // e = rm*64 + kk with kk fastest across tid -> coalesced global reads.
    for (int e = tid; e < TM * TK; e += NTHR) {
      const int rm = e >> 6;
      const int kk = e & (TK - 1);
      const int k  = k0 + kk;
      const int r  = m0 + rm;
      float v = 0.f;
      if (r < M && k < K) {
        int b, vv;
        if (V == 1) { b = r; vv = 0; }
        else        { b = r / V; vv = r - b * V; }
        int s, c;
        if (cshift >= 0) { s = k >> cshift; c = k & (C - 1); }
        else             { s = k / C;       c = k - s * C;   }
        const int sv = idx ? idx[vv * S + s] : vv;
        v = x[((long)b * Vin + sv) * C + c];
      }
      ldsA[rm * LDA + kk] = (_Float16)v;
    }

    if (wave == 0) __builtin_amdgcn_s_wait_tensorcnt(0);
    __syncthreads();

    // ---- 2 k-steps x 2 n-subtiles of v_wmma_f32_16x16x32_f16
    const int arow = (wave * 16 + l15) * LDA;
#pragma unroll
    for (int ks = 0; ks < TK; ks += 32) {
      // A frag per ISA 7.12.2: lane<16 K {0..7,16..23}; lane>=16 K {8..15,24..31}
      const _Float16* pA = &ldsA[arow + ks + half * 8];
      const v8h alo = *(const v8h*)(pA);
      const v8h ahi = *(const v8h*)(pA + 16);
      v16h a;
#pragma unroll
      for (int e = 0; e < 8; ++e) { a[e] = alo[e]; a[e + 8] = ahi[e]; }

#pragma unroll
      for (int ns = 0; ns < 2; ++ns) {
        // B frag: lane<16 holds col n with K {0..7,16..23}; lane>=16 with K {8..15,24..31}
        const _Float16* pB = &ldsB[(ns * 16 + l15) * LDB + ks + half * 8];
        const v8h blo = *(const v8h*)(pB);
        const v8h bhi = *(const v8h*)(pB + 16);
        v16h b;
#pragma unroll
        for (int e = 0; e < 8; ++e) { b[e] = blo[e]; b[e + 8] = bhi[e]; }

        if (ns == 0)
          acc0 = __builtin_amdgcn_wmma_f32_16x16x32_f16(
              false, a, false, b, (short)0, acc0, false, false);
        else
          acc1 = __builtin_amdgcn_wmma_f32_16x16x32_f16(
              false, a, false, b, (short)0, acc1, false, false);
      }
    }
    __syncthreads();
  }

  // ---- epilogue: C/D layout = lane n (mod 16), VGPR i -> row i (+8 lanes>=16)
#pragma unroll
  for (int ns = 0; ns < 2; ++ns) {
    const int   o  = n0 + ns * 16 + l15;
    const float bv = (o < O) ? bias[o] : 0.f;
    const v8f&  ac = ns ? acc1 : acc0;
#pragma unroll
    for (int i = 0; i < 8; ++i) {
      const int r = m0 + wave * 16 + (half ? 8 : 0) + i;
      if (r < M && o < O) {
        float v = ac[i] + bv;
        if (act) v = (v > 0.f) ? v : (__expf(v) - 1.f);
        out[(long)r * O + o] = v;
      }
    }
  }
}

// ---------------------------------------------------------------------------
// Sparse vertex pool: out[b, row[i], c] += val[i] * x[b, col[i], c]
// One thread per (i, b, c); c fastest -> coalesced reads, float atomics out.
// ---------------------------------------------------------------------------
__global__ __launch_bounds__(256)
void pool_scatter(const float* __restrict__ x, const int* __restrict__ row,
                  const int* __restrict__ col, const float* __restrict__ val,
                  float* __restrict__ out,
                  int nnz, int Bn, int C, int Vin, int Vout)
{
  const long gid   = (long)blockIdx.x * blockDim.x + threadIdx.x;
  const long total = (long)nnz * Bn * C;
  if (gid >= total) return;
  const int c = (int)(gid % C);
  const long t = gid / C;
  const int b = (int)(t % Bn);
  const int i = (int)(t / Bn);
  const float v = val[i] * x[((long)b * Vin + col[i]) * C + c];
  atomicAdd(&out[((long)b * Vout + row[i]) * C + c], v);
}

// ---------------------------------------------------------------------------
extern "C" void kernel_launch(void* const* d_in, const int* in_sizes, int n_in,
                              void* d_out, int out_size, void* d_ws, size_t ws_size,
                              hipStream_t stream) {
  (void)in_sizes; (void)n_in; (void)out_size; (void)ws_size;

  static const int Lv[5] = {20000, 5000, 1250, 312, 78};
  const int B = 16, S = 12;

  const float* x = (const float*)d_in[0];
  const int* sp[4];
  for (int i = 0; i < 4; ++i) sp[i] = (const int*)d_in[1 + i];
  const int *drow[4], *dcol[4], *urow[4], *ucol[4];
  const float *dval[4], *uval[4];
  for (int i = 0; i < 4; ++i) {
    const int base = 5 + 6 * i;
    drow[i] = (const int*)  d_in[base + 0];
    dcol[i] = (const int*)  d_in[base + 1];
    dval[i] = (const float*)d_in[base + 2];
    urow[i] = (const int*)  d_in[base + 3];
    ucol[i] = (const int*)  d_in[base + 4];
    uval[i] = (const float*)d_in[base + 5];
  }

  // scratch layout: buf0 (81.92MB) | buf1 (40.96MB) | f16 weight arena (~14MB)
  float*     buf0   = (float*)d_ws;
  float*     buf1   = (float*)((char*)d_ws + 81920000ull);
  _Float16*  warena = (_Float16*)((char*)d_ws + 122880000ull);
  size_t     woff   = 0;

  struct WH { const _Float16* p; int Opad; int Kpad; };

  auto cvt = [&](int widx, int O, int K) -> WH {
    const float* W = (const float*)d_in[widx];
    const int Opad = ((O + TN - 1) / TN) * TN;
    const int Kpad = ((K + TK - 1) / TK) * TK;
    _Float16* dst = warena + woff;
    woff += (size_t)Opad * Kpad;
    const int total = Opad * Kpad;
    cvt_pad_f16<<<dim3((total + 255) / 256), dim3(256), 0, stream>>>(
        W, dst, O, K, Kpad, total);
    return {dst, Opad, Kpad};
  };

  // pre-convert all weights once (re-run each call; ws is not preserved state)
  WH We[4], Wd[4];
  const int encK[4] = {3 * S, 32 * S, 64 * S, 128 * S};
  const int encO[4] = {32, 64, 128, 256};
  for (int i = 0; i < 4; ++i) We[i] = cvt(29 + 2 * i, encO[i], encK[i]);
  const int KF = Lv[4] * 256;  // 19968
  WH Wenc = cvt(37, 128, KF);
  WH Wdec = cvt(39, KF, 128);
  const int decK[4] = {256 * S, 256 * S, 128 * S, 64 * S};
  const int decO[4] = {256, 128, 64, 32};
  for (int i = 0; i < 4; ++i) Wd[i] = cvt(41 + 2 * i, decO[i], decK[i]);
  WH Wout = cvt(49, 3, 32 * S);

  auto log2p2 = [](int c) -> int {   // log2(c) if power of two, else -1
    return (c > 0 && (c & (c - 1)) == 0) ? (31 - __builtin_clz((unsigned)c)) : -1;
  };

  auto conv = [&](const float* in, const int* idx, const WH& w, int bidx,
                  float* op, int V, int Vin, int C, int O, int act, int Sloc) {
    const float* bp = (const float*)d_in[bidx];
    const int M = B * V;
    dim3 grid((M + TM - 1) / TM, w.Opad / TN);
    sconv_wmma<<<grid, dim3(NTHR), 0, stream>>>(in, idx, w.p, bp, op,
                                                B, V, Vin, Sloc, C, O, act,
                                                log2p2(C), w.Kpad);
  };
  auto pool = [&](const float* in, const int* r, const int* c, const float* v,
                  float* op, int nnz, int C, int Vin, int Vout) {
    hipMemsetAsync(op, 0, (size_t)B * Vout * C * sizeof(float), stream);
    const long total = (long)nnz * B * C;
    const int blocks = (int)((total + 255) / 256);
    pool_scatter<<<dim3(blocks), dim3(256), 0, stream>>>(in, r, c, v, op,
                                                         nnz, B, C, Vin, Vout);
  };

  // ---------------- encoder ----------------
  conv(x,    sp[0], We[0], 30, buf0, Lv[0], Lv[0],   3,  32, 1, S);
  pool(buf0, drow[0], dcol[0], dval[0], buf1, 3 * Lv[1],  32, Lv[0], Lv[1]);
  conv(buf1, sp[1], We[1], 32, buf0, Lv[1], Lv[1],  32,  64, 1, S);
  pool(buf0, drow[1], dcol[1], dval[1], buf1, 3 * Lv[2],  64, Lv[1], Lv[2]);
  conv(buf1, sp[2], We[2], 34, buf0, Lv[2], Lv[2],  64, 128, 1, S);
  pool(buf0, drow[2], dcol[2], dval[2], buf1, 3 * Lv[3], 128, Lv[2], Lv[3]);
  conv(buf1, sp[3], We[3], 36, buf0, Lv[3], Lv[3], 128, 256, 1, S);
  pool(buf0, drow[3], dcol[3], dval[3], buf1, 3 * Lv[4], 256, Lv[3], Lv[4]);

  // latent FCs (dense GEMM via same kernel, idx=null, V=1, C=K, S=1)
  conv(buf1, nullptr, Wenc, 38, buf0, 1, 1, KF, 128, 0, 1);  // z    [16,128]
  conv(buf0, nullptr, Wdec, 40, buf1, 1, 1, 128, KF, 0, 1);  // hdec [16,78,256]

  // ---------------- decoder ----------------
  pool(buf1, urow[3], ucol[3], uval[3], buf0, 3 * Lv[3], 256, Lv[4], Lv[3]);
  conv(buf0, sp[3], Wd[0], 42, buf1, Lv[3], Lv[3], 256, 256, 1, S);
  pool(buf1, urow[2], ucol[2], uval[2], buf0, 3 * Lv[2], 256, Lv[3], Lv[2]);
  conv(buf0, sp[2], Wd[1], 44, buf1, Lv[2], Lv[2], 256, 128, 1, S);
  pool(buf1, urow[1], ucol[1], uval[1], buf0, 3 * Lv[1], 128, Lv[2], Lv[1]);
  conv(buf0, sp[1], Wd[2], 46, buf1, Lv[1], Lv[1], 128,  64, 1, S);
  pool(buf1, urow[0], ucol[0], uval[0], buf0, 3 * Lv[0],  64, Lv[1], Lv[0]);
  conv(buf0, sp[0], Wd[3], 48, buf1, Lv[0], Lv[0],  64,  32, 1, S);

  // output spiral conv (O=3, no activation) -> d_out [16, 20000, 3]
  conv(buf1, sp[0], Wout, 50, (float*)d_out, Lv[0], Lv[0], 32, 3, 0, S);
}
